// MinConvExpLSTMCell_67542655697030
// MI455X (gfx1250) — matrix operand order repaired and checked by reading, not compiled, not run on hardware
//
#include <hip/hip_runtime.h>
#include <hip/hip_bf16.h>
#include <cstdint>
#include <cstddef>

// ---------------------------------------------------------------------------
// Problem constants (from reference)
// ---------------------------------------------------------------------------
#define BB    4
#define TT    64
#define IMGS  (BB * TT)      // 256
#define HID   64
#define HT    32
#define WD    32
#define NPOS  (HT * WD)      // 1024 spatial positions per image
#define CO    (3 * HID)      // 192 conv output channels
#define CI    HID            // 64 conv input channels

typedef __bf16 bf16_t;
typedef __attribute__((ext_vector_type(16))) __bf16 v16bf;
typedef __attribute__((ext_vector_type(8)))  __bf16 v8bf;
typedef __attribute__((ext_vector_type(8)))  float  v8f;

__device__ __forceinline__ bf16_t f2bf(float f) {
    unsigned u = __builtin_bit_cast(unsigned, f);
    u += 0x7FFFu + ((u >> 16) & 1u);          // round-to-nearest-even
    unsigned short h = (unsigned short)(u >> 16);
    return __builtin_bit_cast(bf16_t, h);
}

// CDNA5 async global->LDS copy, 16 bytes per lane (tracked by ASYNCcnt).
__device__ __forceinline__ void async_copy16(void* lds_dst, const void* gsrc) {
    uint32_t l = (uint32_t)(uintptr_t)lds_dst;      // LDS aperture low bits = LDS byte offset
    asm volatile("global_load_async_to_lds_b128 %0, %1, off"
                 :: "v"(l), "v"(gsrc)
                 : "memory");
}
__device__ __forceinline__ void wait_async0() {
    asm volatile("s_wait_asynccnt 0" ::: "memory");
}

// ---------------------------------------------------------------------------
// Kernel 1: pack conv weights f32 [co][ci][3][3] -> bf16 Wt[rs][co][ci]
// ---------------------------------------------------------------------------
__global__ void pack_weights(const float* __restrict__ conv_w,
                             bf16_t* __restrict__ Wt) {
    int idx = blockIdx.x * blockDim.x + threadIdx.x;   // 9*192*64 = 110592
    if (idx >= 9 * CO * CI) return;
    int ci = idx & (CI - 1);
    int co = (idx >> 6) % CO;
    int rs = idx / (CO * CI);
    int r = rs / 3, s = rs % 3;
    Wt[idx] = f2bf(conv_w[((co * CI + ci) * 3 + r) * 3 + s]);
}

// ---------------------------------------------------------------------------
// Kernel 2: transpose x f32 NCHW -> bf16 NHWC  (Xb[img][h][w][ci])
// grid: (NPOS/32, IMGS), block: 256
// ---------------------------------------------------------------------------
__global__ __launch_bounds__(256) void pack_x_nhwc(const float* __restrict__ x,
                                                   bf16_t* __restrict__ Xb) {
    __shared__ float tile[CI][33];
    int img = blockIdx.y;
    int n0  = blockIdx.x * 32;
    int tx  = threadIdx.x & 31;
    int ty  = threadIdx.x >> 5;                    // 0..7
    const float* xp = x + (size_t)img * CI * NPOS;
    #pragma unroll
    for (int cc = 0; cc < 8; ++cc) {
        int ci = cc * 8 + ty;
        tile[ci][tx] = xp[(size_t)ci * NPOS + n0 + tx];
    }
    __syncthreads();
    unsigned* out = (unsigned*)Xb;
    #pragma unroll
    for (int p = 0; p < 4; ++p) {
        int n = p * 8 + ty;
        unsigned short h0 = __builtin_bit_cast(unsigned short, f2bf(tile[tx * 2 + 0][n]));
        unsigned short h1 = __builtin_bit_cast(unsigned short, f2bf(tile[tx * 2 + 1][n]));
        out[((size_t)img * NPOS + n0 + n) * (CI / 2) + tx] = (unsigned)h0 | ((unsigned)h1 << 16);
    }
}

// ---------------------------------------------------------------------------
// Kernel 3: block-tiled implicit-GEMM 3x3 conv, v_wmma_f32_16x16x32_bf16.
// Block (8 waves) computes 32 co x 256 positions (8 rows x 32 cols) of one
// image.  X halo tile (10 x 34 x 64ci, edges pre-zeroed) and the 32-co weight
// slab live in LDS (~80 KB), filled with async global->LDS b128 copies.
// Hot loop: unconditional ds_load -> WMMA, 72 WMMAs per wave (4 accumulators,
// A fragment reused 4x).  grid.x = IMGS * (CO/32) * (NPOS/256) = 6144.
// ---------------------------------------------------------------------------
union AFrag { v16bf v; v8bf h[2]; };

#define XROWS 10
#define XCOLS 34

__global__ __launch_bounds__(256) void conv3x3_wmma(
        const bf16_t* __restrict__ Xb,
        const bf16_t* __restrict__ Wt,
        const float*  __restrict__ conv_b,
        float* __restrict__ y) {
    __shared__ bf16_t Xl[XROWS * XCOLS * CI];      // 43,520 B
    __shared__ bf16_t Wl[9 * 32 * CI];             // 36,864 B

    int bid  = blockIdx.x;
    int img  = bid / 24;
    int rem  = bid % 24;
    int cob  = rem >> 2;                           // 0..5 : co block of 32
    int nblk = rem & 3;                            // 0..3 : 8-row strip
    int h0   = nblk * 8;

    int tid  = threadIdx.x;
    int lane = tid & 31;
    int wave = tid >> 5;

    // ---- zero the halo columns (lx = 0 and 33) for all 10 rows ----
    for (int i = tid; i < XROWS * 2 * (CI / 8); i += 256) {   // 160 chunks
        int ly   = i >> 4;
        int side = (i >> 3) & 1;
        int ci0  = (i & 7) * 8;
        int lx   = side ? (XCOLS - 1) : 0;
        *(v8bf*)(Xl + ((size_t)ly * XCOLS + lx) * CI + ci0) = (v8bf){};
    }

    // ---- async-fill interior X rows (zero-store rows outside the image) ----
    for (int i = tid; i < XROWS * 256; i += 256) { // 2560 16B chunks
        int ly  = i >> 8;                          // uniform per wave
        int c   = i & 255;
        int pos = c >> 3;                          // 0..31
        int ci0 = (c & 7) * 8;
        bf16_t* dst = Xl + ((size_t)ly * XCOLS + 1 + pos) * CI + ci0;
        int gh = h0 + ly - 1;
        if ((unsigned)gh < HT) {
            const bf16_t* src = Xb + (((size_t)img * HT + gh) * WD + pos) * CI + ci0;
            async_copy16(dst, src);
        } else {
            *(v8bf*)dst = (v8bf){};
        }
    }

    // ---- async-fill the 32-co weight slab: Wl[rs][col][ci] ----
    for (int i = tid; i < 9 * 32 * (CI / 8); i += 256) {      // 2304 chunks
        int rs  = i >> 8;
        int c   = i & 255;
        int col = c >> 3;
        int ci0 = (c & 7) * 8;
        async_copy16(Wl + ((size_t)rs * 32 + col) * CI + ci0,
                     Wt + ((size_t)(rs * CO + cob * 32 + col)) * CI + ci0);
    }

    wait_async0();
    __syncthreads();

    // ---- wave tiling: 2 waves along co (16 each) x 4 waves along n (64 each)
    int wco  = wave & 1;
    int wn   = wave >> 1;
    int lm   = lane & 15;
    int koff = lane >> 4;

    v8f c0 = {}, c1 = {}, c2 = {}, c3 = {};
    const bf16_t* wrowbase = Wl + (size_t)(wco * 16 + lm) * CI;

    // local rows/cols of this wave's 4 n-tiles (position = wn*64 + t*16 + lm)
    int lr[4], lc[4];
    #pragma unroll
    for (int t = 0; t < 4; ++t) {
        int nloc = wn * 64 + t * 16 + lm;
        lr[t] = nloc >> 5;
        lc[t] = nloc & 31;
    }

    #pragma unroll
    for (int rs = 0; rs < 9; ++rs) {
        int r = rs / 3, s = rs % 3;
        const bf16_t* wrs = wrowbase + (size_t)rs * 32 * CI;
        #pragma unroll
        for (int cik = 0; cik < 2; ++cik) {
            int cb = cik * 32;
            // A fragment (contiguous k order: k = 16*half + s, matching B)
            AFrag a;
            a.h[0] = *(const v8bf*)(wrs + cb + koff * 16);
            a.h[1] = *(const v8bf*)(wrs + cb + koff * 16 + 8);
            // 4 B fragments from the LDS halo tile, unconditional
            #pragma unroll
            for (int t = 0; t < 4; ++t) {
                const bf16_t* xp = Xl + ((size_t)(lr[t] + r) * XCOLS + (lc[t] + s)) * CI
                                     + cb + koff * 16;
                v16bf b = *(const v16bf*)xp;
                v8f& acc = (t == 0) ? c0 : (t == 1) ? c1 : (t == 2) ? c2 : c3;
                acc = __builtin_amdgcn_wmma_f32_16x16x32_bf16(
                          false, a.v, false, b, (short)0, acc, false, false);
            }
        }
    }

    // ---- epilogue: bias + store fp32 gates y[img][co][n] ----
    float bias[8];
    #pragma unroll
    for (int v = 0; v < 8; ++v)
        bias[v] = conv_b[cob * 32 + wco * 16 + v + koff * 8];

    float* yp = y + (size_t)img * CO * NPOS + (size_t)nblk * 256;
    #pragma unroll
    for (int t = 0; t < 4; ++t) {
        v8f acc = (t == 0) ? c0 : (t == 1) ? c1 : (t == 2) ? c2 : c3;
        int nn = wn * 64 + t * 16 + lm;
        #pragma unroll
        for (int v = 0; v < 8; ++v) {
            int co = cob * 32 + wco * 16 + v + koff * 8;
            yp[(size_t)co * NPOS + nn] = acc[v] + bias[v];
        }
    }
}

// ---------------------------------------------------------------------------
// Kernel 4: log-space recurrence over T per (b, hid, h, w) sequence
// ---------------------------------------------------------------------------
__device__ __forceinline__ float log_sigmoid_f(float xv) {
    return fminf(xv, 0.0f) - log1pf(expf(-fabsf(xv)));
}
__device__ __forceinline__ float log_g_f(float xv) {
    return (xv >= 0.0f) ? logf(xv + 0.5f) : log_sigmoid_f(xv);
}
__device__ __forceinline__ float logaddexp_f(float a, float b) {
    return fmaxf(a, b) + log1pf(expf(-fabsf(a - b)));
}

__global__ __launch_bounds__(256) void scan_time(const float* __restrict__ y,
                                                 const float* __restrict__ h0,
                                                 float* __restrict__ out) {
    int seq = blockIdx.x * blockDim.x + threadIdx.x;   // B*HID*NPOS = 262144
    if (seq >= BB * HID * NPOS) return;
    int n   = seq & (NPOS - 1);
    int hid = (seq >> 10) & (HID - 1);
    int b   = seq >> 16;

    float L = log_g_f(h0[((size_t)b * HID + hid) * NPOS + n]);

    for (int t = 0; t < TT; ++t) {
        size_t base = ((size_t)(b * TT + t)) * CO * NPOS;
        float fg = y[base + (size_t)hid * NPOS + n];
        float ig = y[base + (size_t)(HID + hid) * NPOS + n];
        float ht = y[base + (size_t)(2 * HID + hid) * NPOS + n];
        float diff  = fg - ig;
        float log_f = log_sigmoid_f(diff);
        float log_i = log_f - diff;
        float lv    = log_i + log_g_f(ht);
        L = logaddexp_f(log_f + L, lv);
        out[((size_t)(b * TT + t) * HID + hid) * NPOS + n] = expf(L);
    }
}

// ---------------------------------------------------------------------------
// Launcher
// ---------------------------------------------------------------------------
extern "C" void kernel_launch(void* const* d_in, const int* in_sizes, int n_in,
                              void* d_out, int out_size, void* d_ws, size_t ws_size,
                              hipStream_t stream) {
    const float* x      = (const float*)d_in[0];   // (256, 64, 32, 32)
    const float* conv_w = (const float*)d_in[1];   // (192, 64, 3, 3)
    const float* conv_b = (const float*)d_in[2];   // (192,)
    const float* h0     = (const float*)d_in[3];   // (4, 64, 32, 32)
    float* out          = (float*)d_out;

    char* ws = (char*)d_ws;
    const size_t xb_bytes = (size_t)IMGS * NPOS * CI * sizeof(bf16_t);   // 32 MiB
    const size_t wt_bytes = (size_t)9 * CO * CI * sizeof(bf16_t);        // 216 KiB
    const size_t wt_off   = xb_bytes;
    const size_t y_off    = xb_bytes + ((wt_bytes + 255) & ~(size_t)255);

    bf16_t* Xb = (bf16_t*)ws;
    bf16_t* Wt = (bf16_t*)(ws + wt_off);
    float*  y  = (float*)(ws + y_off);             // 192 MiB fp32 gate buffer

    pack_weights<<<(9 * CO * CI + 255) / 256, 256, 0, stream>>>(conv_w, Wt);
    pack_x_nhwc<<<dim3(NPOS / 32, IMGS), 256, 0, stream>>>(x, Xb);
    conv3x3_wmma<<<IMGS * (CO / 32) * (NPOS / 256), 256, 0, stream>>>(Xb, Wt, conv_b, y);
    scan_time<<<(BB * HID * NPOS) / 256, 256, 0, stream>>>(y, h0, out);
}
